// SparseGatingNetwork_54451595378909
// MI455X (gfx1250) — matrix operand order repaired and checked by reading, not compiled, not run on hardware
//
#include <hip/hip_runtime.h>
#include <stdint.h>

typedef __attribute__((ext_vector_type(16))) __bf16 v16bf;
typedef __attribute__((ext_vector_type(8)))  float  v8f;

namespace {
constexpr int Dk        = 4096;   // feature dim
constexpr int Ex        = 64;     // experts
constexpr int Nrows     = 65536;  // tokens
constexpr int RowsPerWG = 128;    // 8 waves * 16 rows
constexpr int WavesPerWG = 8;
}

// Truncation split of two fp32 into packed bf16 hi (exact top 16 bits) and
// packed bf16 lo (truncated remainder). x ~= hi + lo with |err| <= 2^-16 |x|.
__device__ __forceinline__ void split2(float f0, float f1, uint32_t& hi, uint32_t& lo) {
  uint32_t u0 = __float_as_uint(f0), u1 = __float_as_uint(f1);
  uint32_t h0 = u0 & 0xFFFF0000u,   h1 = u1 & 0xFFFF0000u;
  hi = (h0 >> 16) | h1;
  float l0 = f0 - __uint_as_float(h0);
  float l1 = f1 - __uint_as_float(h1);
  lo = (__float_as_uint(l0) >> 16) | (__float_as_uint(l1) & 0xFFFF0000u);
}

// Pre-pass: split W [E*D] fp32 into bf16 Whi/Wlo (row-major, k-contiguous).
__global__ void wsplit_kernel(const float4* __restrict__ W4,
                              uint2* __restrict__ hi2, uint2* __restrict__ lo2) {
  int i = blockIdx.x * blockDim.x + threadIdx.x;  // one float4 (4 elems)
  float4 f = W4[i];
  uint2 h, l;
  split2(f.x, f.y, h.x, l.x);
  split2(f.z, f.w, h.y, l.y);
  hi2[i] = h;
  lo2[i] = l;
}

union BFrag { v16bf v; uint4 q[2]; uint32_t u[8]; };

__global__ __launch_bounds__(256) void gating_kernel(
    const float*    __restrict__ x,
    const float*    __restrict__ bias,
    const uint16_t* __restrict__ Whi,   // bf16 bits [E*D]
    const uint16_t* __restrict__ Wlo,   // bf16 bits [E*D]
    float* __restrict__ out_w,
    int*   __restrict__ out_i)
{
  __shared__ float sm[WavesPerWG][16][65];   // 65 -> conflict-free column scan

  const int lane = threadIdx.x & 31;
  const int wave = threadIdx.x >> 5;
  const int half = lane >> 4;   // 0: lanes 0-15, 1: lanes 16-31
  const int lq   = lane & 15;

  const int row = blockIdx.x * RowsPerWG + wave * 16 + lq;
  const float* __restrict__ xrow = x + (size_t)row * Dk;

  // B fragment base for expert (t*16 + lq); half selects K {0..7,16..23} vs {8..15,24..31}.
  const uint16_t* __restrict__ wh = Whi + (size_t)lq * Dk + half * 8;
  const uint16_t* __restrict__ wl = Wlo + (size_t)lq * Dk + half * 8;

  v8f acc[4] = {v8f{}, v8f{}, v8f{}, v8f{}};

  for (int kb = 0; kb < Dk; kb += 32) {
    if (kb + 64 <= Dk) {
      __builtin_prefetch(xrow + kb + 32 + half * 8, 0, 0);   // global_prefetch_b8
    }
    // A fragment, 16-bit A 16x32 layout:
    //   half 0: M=lq, K in {0..7} u {16..23}; half 1: M=lq, K in {8..15} u {24..31}
    const int g1 = kb + half * 8;
    float4 a0 = *(const float4*)(xrow + g1);
    float4 a1 = *(const float4*)(xrow + g1 + 4);
    float4 a2 = *(const float4*)(xrow + g1 + 16);
    float4 a3 = *(const float4*)(xrow + g1 + 20);

    BFrag Ahi, Alo;
    split2(a0.x, a0.y, Ahi.u[0], Alo.u[0]);
    split2(a0.z, a0.w, Ahi.u[1], Alo.u[1]);
    split2(a1.x, a1.y, Ahi.u[2], Alo.u[2]);
    split2(a1.z, a1.w, Ahi.u[3], Alo.u[3]);
    split2(a2.x, a2.y, Ahi.u[4], Alo.u[4]);
    split2(a2.z, a2.w, Ahi.u[5], Alo.u[5]);
    split2(a3.x, a3.y, Ahi.u[6], Alo.u[6]);
    split2(a3.z, a3.w, Ahi.u[7], Alo.u[7]);

#pragma unroll
    for (int t = 0; t < 4; ++t) {
      // B 32x16 bf16 fragment: lane -> column N = lq, K = half*8 + {0..7} u {16..23}.
      const uint4* ph = (const uint4*)(wh + (size_t)(t * 16) * Dk + kb);
      const uint4* pl = (const uint4*)(wl + (size_t)(t * 16) * Dk + kb);
      BFrag Bh, Bl;
      Bh.q[0] = ph[0]; Bh.q[1] = ph[2];
      Bl.q[0] = pl[0]; Bl.q[1] = pl[2];

      acc[t] = __builtin_amdgcn_wmma_f32_16x16x32_bf16(
          false, Ahi.v, false, Bh.v, (short)0, acc[t], false, false);
      acc[t] = __builtin_amdgcn_wmma_f32_16x16x32_bf16(
          false, Ahi.v, false, Bl.v, (short)0, acc[t], false, false);
      acc[t] = __builtin_amdgcn_wmma_f32_16x16x32_bf16(
          false, Alo.v, false, Bh.v, (short)0, acc[t], false, false);
    }
  }

  // Epilogue: bias + spill logits to LDS. C/D layout: VGPR r -> M = r + half*8,
  // N = lq, for expert tile t -> column t*16 + lq.
#pragma unroll
  for (int t = 0; t < 4; ++t) {
    float bv = bias[t * 16 + lq];
#pragma unroll
    for (int r = 0; r < 8; ++r) {
      sm[wave][r + half * 8][t * 16 + lq] = acc[t][r] + bv;
    }
  }
  __syncthreads();

  // Lanes 0-15: per-row top-2 + stable softmax over 64 experts.
  if (lane < 16) {
    const float* rowp = &sm[wave][lane][0];
    float m1 = -__builtin_inff(), m2 = -__builtin_inff();
    int i1 = 0, i2 = 0;
    for (int j = 0; j < Ex; ++j) {
      float v = rowp[j];
      if (v > m1)      { m2 = m1; i2 = i1; m1 = v; i1 = j; }
      else if (v > m2) { m2 = v;  i2 = j; }
    }
    float s = 0.0f;
    for (int j = 0; j < Ex; ++j) s += expf(rowp[j] - m1);

    const int gr  = blockIdx.x * RowsPerWG + wave * 16 + lane;
    const float inv = 1.0f / s;
    out_w[gr * 2 + 0] = inv;                  // exp(m1-m1)/s
    out_w[gr * 2 + 1] = expf(m2 - m1) * inv;
    out_i[gr * 2 + 0] = i1;
    out_i[gr * 2 + 1] = i2;
  }
}

extern "C" void kernel_launch(void* const* d_in, const int* in_sizes, int n_in,
                              void* d_out, int out_size, void* d_ws, size_t ws_size,
                              hipStream_t stream) {
  const float* x = (const float*)d_in[0];   // [N, D] fp32
  const float* W = (const float*)d_in[1];   // [E, D] fp32
  const float* b = (const float*)d_in[2];   // [E]    fp32

  // Workspace: Whi (E*D bf16 = 512 KB) then Wlo (512 KB).
  uint16_t* Whi = (uint16_t*)d_ws;
  uint16_t* Wlo = Whi + (size_t)Ex * Dk;

  float* out_w = (float*)d_out;                       // [N, 2] fp32 weights
  int*   out_i = (int*)d_out + (size_t)Nrows * 2;     // [N, 2] i32 indices

  // 1) Split W into bf16 hi/lo (one float4 per thread).
  const int welems4 = Ex * Dk / 4;                    // 65536
  wsplit_kernel<<<welems4 / 256, 256, 0, stream>>>(
      (const float4*)W, (uint2*)Whi, (uint2*)Wlo);

  // 2) Fused GEMM + softmax + top-2.
  gating_kernel<<<Nrows / RowsPerWG, 256, 0, stream>>>(
      x, b, Whi, Wlo, out_w, out_i);
}
